// TransformerEncoderMoE_30459908063737
// MI455X (gfx1250) — matrix operand order repaired and checked by reading, not compiled, not run on hardware
//
#include <hip/hip_runtime.h>
#include <hip/hip_bf16.h>
#include <cstdint>

// ---------------------------------------------------------------------------
// TransformerEncoderMoE for MI455X (gfx1250, wave32, WMMA bf16 path,
// async global->LDS staging).
//
// Input flattening assumption (setup_inputs dict order, recursive):
//   0: input_ids [B,S] i32      1: attention_mask [B,S] i32
//   2: tok [V,H] f32            3: pos [MAXS,H] f32
//   4: g_f [H]                  5: b_f [H]
//   6 + l*18 + j : layer l tensors in order
//        g1,b1,Wq,bq,Wk,bk,Wv,bv,Wo,bo,g2,b2,Wg,bg,W1,b1e,W2,b2e
// Output: final LN(x) [B,S,H] f32.
// ---------------------------------------------------------------------------

typedef __bf16 bf16;
typedef __attribute__((ext_vector_type(16))) __bf16 v16bf;
typedef __attribute__((ext_vector_type(8)))  float  v8f;

namespace cfg {
constexpr int Bsz   = 8;
constexpr int Seq   = 1024;
constexpr int Hdim  = 1024;
constexpr int NHead = 16;
constexpr int HeadD = 64;
constexpr int FF    = 2048;
constexpr int NE    = 8;
constexpr int NL    = 4;
constexpr int TOK   = Bsz * Seq;          // 8192 tokens
constexpr int ROWCAP = TOK * 2 + NE * 16; // top-2 routing rows, padded per expert
}

__device__ __forceinline__ v8f wmma_bf16(v16bf a, v16bf b, v8f c) {
  // v_wmma_f32_16x16x32_bf16  D = A(16x32) * B(32x16) + C(16x16 f32)
  return __builtin_amdgcn_wmma_f32_16x16x32_bf16(false, a, false, b, (short)0, c,
                                                 false, false);
}

// Async DMA of one 16-byte chunk per lane: global -> LDS, tracked by ASYNCcnt.
// (CDNA5 GLOBAL_LOAD_ASYNC_TO_LDS_B128; generic LDS pointer's low 32 bits are
//  the LDS byte offset per the flat-aperture rules.)
__device__ __forceinline__ void async_copy_b128(void* lds_dst, const void* gsrc) {
  const uint32_t ldsa = (uint32_t)(uintptr_t)lds_dst;
  const uint64_t ga = (uint64_t)(uintptr_t)gsrc;
  asm volatile("global_load_async_to_lds_b128 %0, %1, off"
               :: "v"(ldsa), "v"(ga)
               : "memory");
}
__device__ __forceinline__ void wait_async0() {
  asm volatile("s_wait_asynccnt 0" ::: "memory");
}

// A-fragment (16x32 bf16): lane<16 -> K 0..7 / 16..23; lane>=16 -> K 8..15 / 24..31
__device__ __forceinline__ v16bf load_afrag(const bf16* __restrict__ row, int hf) {
  v16bf a;
  const bf16* p = row + hf * 8;
#pragma unroll
  for (int i = 0; i < 8; ++i) { a[i] = p[i]; a[8 + i] = p[16 + i]; }
  return a;
}

// B-fragment (32x16 bf16), source stored [N][K]: lane<16 -> K 0..7 / 16..23; lane>=16 -> K 8..15 / 24..31
__device__ __forceinline__ v16bf load_bfrag(const bf16* __restrict__ col, int hf) {
  v16bf b;
  const bf16* p = col + hf * 8;
#pragma unroll
  for (int i = 0; i < 8; ++i) { b[i] = p[i]; b[8 + i] = p[16 + i]; }
  return b;
}

// ---------------------------------------------------------------------------
// Elementwise / setup kernels
// ---------------------------------------------------------------------------

__global__ __launch_bounds__(256) void cvt_kernel(const float* __restrict__ in,
                                                  bf16* __restrict__ out, int n) {
  int i = blockIdx.x * 256 + threadIdx.x;
  if (i < n) out[i] = (bf16)in[i];
}

__global__ __launch_bounds__(256) void embed_kernel(const int* __restrict__ ids,
                                                    const float* __restrict__ tokW,
                                                    const float* __restrict__ posW,
                                                    float* __restrict__ x) {
  using namespace cfg;
  const int t = blockIdx.x;
  const int id = ids[t];
  const int sp = t & (Seq - 1);
  for (int i = threadIdx.x; i < Hdim; i += 256)
    x[(size_t)t * Hdim + i] = tokW[(size_t)id * Hdim + i] + posW[(size_t)sp * Hdim + i];
}

template <bool OUTF, bool OUTB>
__global__ __launch_bounds__(256) void ln_kernel(const float* __restrict__ x,
                                                 const float* __restrict__ g,
                                                 const float* __restrict__ b,
                                                 float* __restrict__ outf,
                                                 bf16* __restrict__ outb) {
  using namespace cfg;
  const int t = blockIdx.x;
  const float* row = x + (size_t)t * Hdim;
  float s = 0.f, ss = 0.f;
  for (int i = threadIdx.x; i < Hdim; i += 256) { float v = row[i]; s += v; ss += v * v; }
#pragma unroll
  for (int o = 16; o > 0; o >>= 1) { s += __shfl_xor(s, o, 32); ss += __shfl_xor(ss, o, 32); }
  __shared__ float rs[8], rss[8];
  if ((threadIdx.x & 31) == 0) { rs[threadIdx.x >> 5] = s; rss[threadIdx.x >> 5] = ss; }
  __syncthreads();
  s = 0.f; ss = 0.f;
#pragma unroll
  for (int w = 0; w < 8; ++w) { s += rs[w]; ss += rss[w]; }
  const float mean = s * (1.f / Hdim);
  const float var = ss * (1.f / Hdim) - mean * mean;
  const float inv = rsqrtf(var + 1e-5f);
  for (int i = threadIdx.x; i < Hdim; i += 256) {
    const float v = (row[i] - mean) * inv * g[i] + b[i];
    if (OUTF) outf[(size_t)t * Hdim + i] = v;
    if (OUTB) outb[(size_t)t * Hdim + i] = (bf16)v;
  }
}

// ---------------------------------------------------------------------------
// Generic dense GEMM:  out[M,N] = A[M,K](bf16) * Bw[K,N](bf16) + bias
// Block: 256 thr = 8 waves in 2x4; wave computes 16(M)x64(N) (A-frag reused 4x).
// Block tile 32(M) x 256(N), BK = 32. A staged via async DMA, B transposed.
// ---------------------------------------------------------------------------
template <bool RELU, bool RESID>
__global__ __launch_bounds__(256) void gemm_main(const bf16* __restrict__ A,
                                                 const bf16* __restrict__ Bw,
                                                 const float* __restrict__ bias,
                                                 float* __restrict__ outF,
                                                 bf16* __restrict__ outB,
                                                 const int N, const int K) {
  constexpr int BM = 32, BN = 256, BK = 32;
  __shared__ bf16 As[BM][BK];
  __shared__ bf16 Bs[BN][BK + 8];  // [N][K], padded vs bank conflicts
  const int tid = threadIdx.x, lane = tid & 31, wid = tid >> 5;
  const int wr = wid >> 2, wc = wid & 3;
  const int hf = lane >> 4, nl = lane & 15;
  const int bm = blockIdx.y * BM, bn = blockIdx.x * BN;
  v8f acc[4];
#pragma unroll
  for (int j = 0; j < 4; ++j) acc[j] = v8f{0.f, 0.f, 0.f, 0.f, 0.f, 0.f, 0.f, 0.f};

  for (int k0 = 0; k0 < K; k0 += BK) {
    __syncthreads();
    if (tid < 128) {  // stage A 32x32 via async DMA (16B per lane)
      const int flat = tid * 8, r = flat >> 5, c = flat & 31;
      async_copy_b128(&As[r][c], A + (size_t)(bm + r) * K + k0 + c);
    }
    {  // stage B 32x256 transposed (32 bf16/thread)
      const int idx = tid * 32, kk = idx >> 8, n0 = idx & 255;
      const bf16* s2 = Bw + (size_t)(k0 + kk) * N + bn + n0;
#pragma unroll
      for (int i = 0; i < 32; ++i) Bs[n0 + i][kk] = s2[i];
    }
    if (k0 + BK < K) __builtin_prefetch(Bw + (size_t)(k0 + BK) * N + bn, 0, 1);
    wait_async0();
    __syncthreads();
    const v16bf a = load_afrag(&As[wr * 16 + nl][0], hf);
#pragma unroll
    for (int j = 0; j < 4; ++j)
      acc[j] = wmma_bf16(a, load_bfrag(&Bs[wc * 64 + j * 16 + nl][0], hf), acc[j]);
  }

#pragma unroll
  for (int j = 0; j < 4; ++j)
#pragma unroll
    for (int r = 0; r < 8; ++r) {
      const int row = bm + wr * 16 + r + hf * 8;  // C layout: M = r (+8 for hi half)
      const int col = bn + wc * 64 + j * 16 + nl;
      float v = acc[j][r] + bias[col];
      if (RELU) v = fmaxf(v, 0.f);
      if (RESID) v += outF[(size_t)row * N + col];
      if (outF) outF[(size_t)row * N + col] = v;
      if (outB) outB[(size_t)row * N + col] = (bf16)v;
    }
}

// ---------------------------------------------------------------------------
// Flash attention: one block per (q-tile of 64 rows, head, batch), 4 waves.
// Wave w owns q rows [w*16, w*16+16). Online softmax, all GEMMs via WMMA.
// Q/K tiles staged with async DMA; V staged transposed manually.
// ---------------------------------------------------------------------------
__global__ __launch_bounds__(128) void attention_kernel(const bf16* __restrict__ q,
                                                        const bf16* __restrict__ k,
                                                        const bf16* __restrict__ v,
                                                        const int* __restrict__ amask,
                                                        bf16* __restrict__ o) {
  using namespace cfg;
  const int qt = blockIdx.x, hh = blockIdx.y, bb = blockIdx.z;
  const int tid = threadIdx.x, lane = tid & 31, wid = tid >> 5;
  const int hf = lane >> 4, nl = lane & 15;

  __shared__ bf16 Qs[64][64];
  __shared__ bf16 Ks[64][64];      // [keypos][d]  (B for QK^T: per-lane col = keypos)
  __shared__ bf16 Vs[64][72];      // [d][keypos]  (B for PV:   per-lane col = d)
  __shared__ bf16 Ps[4][16][64];   // per-wave probs round-trip (C-layout -> A-layout)

  const size_t rowbase = (size_t)bb * Seq;
  const size_t hoff = (size_t)hh * HeadD;

  {  // stage Q (64x64) via async DMA: 512 x 16B chunks, 4 per thread
    const bf16* src0 = q + (rowbase + qt * 64) * Hdim + hoff;
#pragma unroll
    for (int it = 0; it < 4; ++it) {
      const int flat = (tid + it * 128) * 8;
      const int r = flat >> 6, c = flat & 63;
      async_copy_b128(&Qs[r][c], src0 + (size_t)r * Hdim + c);
    }
  }
  wait_async0();
  __syncthreads();
  const v16bf aq0 = load_afrag(&Qs[wid * 16 + nl][0], hf);
  const v16bf aq1 = load_afrag(&Qs[wid * 16 + nl][32], hf);

  v8f oacc[4];
#pragma unroll
  for (int j = 0; j < 4; ++j) oacc[j] = v8f{0.f, 0.f, 0.f, 0.f, 0.f, 0.f, 0.f, 0.f};
  float rmax[8], rsum[8];
#pragma unroll
  for (int r = 0; r < 8; ++r) { rmax[r] = -3.0e38f; rsum[r] = 0.f; }

  for (int kt = 0; kt < Seq; kt += 64) {
    __syncthreads();
    {  // stage K (async DMA) and V transposed (manual)
      const bf16* ksrc = k + (rowbase + kt) * Hdim + hoff;
      const bf16* vsrc = v + (rowbase + kt) * Hdim + hoff;
#pragma unroll
      for (int it = 0; it < 4; ++it) {
        const int flat = (tid + it * 128) * 8;
        const int r = flat >> 6, c = flat & 63;
        async_copy_b128(&Ks[r][c], ksrc + (size_t)r * Hdim + c);
        const bf16* s3 = vsrc + (size_t)r * Hdim + c;
#pragma unroll
        for (int i = 0; i < 8; ++i) Vs[c + i][r] = s3[i];
      }
    }
    if (kt + 64 < Seq) __builtin_prefetch(k + (rowbase + kt + 64) * Hdim + hoff, 0, 1);
    wait_async0();
    __syncthreads();

    // scores S = Q K^T  (K-dim = HeadD = 64 -> two WMMA steps per 16-col subtile)
    v8f sacc[4];
#pragma unroll
    for (int j = 0; j < 4; ++j) {
      sacc[j] = v8f{0.f, 0.f, 0.f, 0.f, 0.f, 0.f, 0.f, 0.f};
      sacc[j] = wmma_bf16(aq0, load_bfrag(&Ks[j * 16 + nl][0], hf), sacc[j]);
      sacc[j] = wmma_bf16(aq1, load_bfrag(&Ks[j * 16 + nl][32], hf), sacc[j]);
    }
    // scale + additive mask (column-dependent)
#pragma unroll
    for (int j = 0; j < 4; ++j) {
      const int kp = kt + j * 16 + nl;
      const float mb = (amask[bb * Seq + kp] == 0) ? -1e9f : 0.f;
#pragma unroll
      for (int r = 0; r < 8; ++r) sacc[j][r] = sacc[j][r] * 0.125f + mb;
    }
    // online softmax: row stats live per VGPR-index r, reduced across 16 lanes
    float nm[8];
#pragma unroll
    for (int r = 0; r < 8; ++r) {
      float m2 = fmaxf(fmaxf(sacc[0][r], sacc[1][r]), fmaxf(sacc[2][r], sacc[3][r]));
#pragma unroll
      for (int off = 8; off > 0; off >>= 1) m2 = fmaxf(m2, __shfl_xor(m2, off, 32));
      nm[r] = fmaxf(rmax[r], m2);
    }
    float radd[8];
#pragma unroll
    for (int r = 0; r < 8; ++r) radd[r] = 0.f;
#pragma unroll
    for (int j = 0; j < 4; ++j)
#pragma unroll
      for (int r = 0; r < 8; ++r) {
        const float pv = __expf(sacc[j][r] - nm[r]);
        sacc[j][r] = pv;
        radd[r] += pv;
      }
#pragma unroll
    for (int r = 0; r < 8; ++r) {
#pragma unroll
      for (int off = 8; off > 0; off >>= 1) radd[r] += __shfl_xor(radd[r], off, 32);
      const float f2 = __expf(rmax[r] - nm[r]);
      rsum[r] = rsum[r] * f2 + radd[r];
      rmax[r] = nm[r];
#pragma unroll
      for (int j = 0; j < 4; ++j) oacc[j][r] *= f2;
    }
    // C-layout -> A-layout through LDS, then O += P V
#pragma unroll
    for (int j = 0; j < 4; ++j)
#pragma unroll
      for (int r = 0; r < 8; ++r) Ps[wid][r + hf * 8][j * 16 + nl] = (bf16)sacc[j][r];
    __syncthreads();
#pragma unroll
    for (int ks2 = 0; ks2 < 2; ++ks2) {
      const v16bf ap = load_afrag(&Ps[wid][nl][ks2 * 32], hf);
#pragma unroll
      for (int j = 0; j < 4; ++j)
        oacc[j] = wmma_bf16(ap, load_bfrag(&Vs[j * 16 + nl][ks2 * 32], hf), oacc[j]);
    }
  }

#pragma unroll
  for (int j = 0; j < 4; ++j)
#pragma unroll
    for (int r = 0; r < 8; ++r) {
      const int qrow = qt * 64 + wid * 16 + r + hf * 8;
      o[(rowbase + qrow) * Hdim + hoff + j * 16 + nl] = (bf16)(oacc[j][r] / rsum[r]);
    }
}

// ---------------------------------------------------------------------------
// MoE gating + routing
// ---------------------------------------------------------------------------
__global__ __launch_bounds__(256) void gate_route(const bf16* __restrict__ hbf,
                                                  const float* __restrict__ Wg,
                                                  const float* __restrict__ bg,
                                                  int* __restrict__ cnt,
                                                  int* __restrict__ toklist,
                                                  float* __restrict__ gatelist) {
  using namespace cfg;
  const int wid = threadIdx.x >> 5, lane = threadIdx.x & 31;
  const int t = blockIdx.x * 8 + wid;
  float acc[NE];
#pragma unroll
  for (int e = 0; e < NE; ++e) acc[e] = 0.f;
  for (int i = lane; i < Hdim; i += 32) {
    const float hv = (float)hbf[(size_t)t * Hdim + i];
    const float* wr = Wg + (size_t)i * NE;
#pragma unroll
    for (int e = 0; e < NE; ++e) acc[e] += hv * wr[e];
  }
#pragma unroll
  for (int e = 0; e < NE; ++e)
#pragma unroll
    for (int off = 16; off > 0; off >>= 1) acc[e] += __shfl_xor(acc[e], off, 32);
  if (lane == 0) {
    float lg[NE];
    float mx = -3e38f;
#pragma unroll
    for (int e = 0; e < NE; ++e) { lg[e] = acc[e] + bg[e]; mx = fmaxf(mx, lg[e]); }
    int e0 = 0;
#pragma unroll
    for (int e = 1; e < NE; ++e) if (lg[e] > lg[e0]) e0 = e;
    int e1 = (e0 == 0) ? 1 : 0;
#pragma unroll
    for (int e = 0; e < NE; ++e) if (e != e0 && lg[e] > lg[e1]) e1 = e;
    const float p0 = __expf(lg[e0] - mx), p1 = __expf(lg[e1] - mx);
    const float inv = 1.f / (p0 + p1);
    const int s0 = atomicAdd(&cnt[e0], 1);
    toklist[e0 * TOK + s0] = t; gatelist[e0 * TOK + s0] = p0 * inv;
    const int s1 = atomicAdd(&cnt[e1], 1);
    toklist[e1 * TOK + s1] = t; gatelist[e1 * TOK + s1] = p1 * inv;
  }
}

__global__ void finalize_route(const int* __restrict__ cnt, int* __restrict__ epad,
                               int* __restrict__ ebase) {
  if (threadIdx.x == 0) {
    int base = 0;
    for (int e = 0; e < cfg::NE; ++e) {
      const int p = (cnt[e] + 15) & ~15;  // pad to WMMA M-tile
      epad[e] = p; ebase[e] = base; base += p;
    }
    ebase[cfg::NE] = base;
  }
}

__global__ __launch_bounds__(256) void scatter_route(const int* __restrict__ cnt,
                                                     const int* __restrict__ epad,
                                                     const int* __restrict__ ebase,
                                                     const int* __restrict__ toklist,
                                                     const float* __restrict__ gatelist,
                                                     int* __restrict__ rowtok,
                                                     float* __restrict__ rowgate) {
  using namespace cfg;
  const int e = blockIdx.y;
  const int s = blockIdx.x * 256 + threadIdx.x;
  if (s >= epad[e]) return;
  const int r = ebase[e] + s;
  if (s < cnt[e]) { rowtok[r] = toklist[e * TOK + s]; rowgate[r] = gatelist[e * TOK + s]; }
  else            { rowtok[r] = 0; rowgate[r] = 0.f; }  // padded row: gate 0
}

// ---------------------------------------------------------------------------
// MoE expert GEMMs. Tile 16(M) x 256(N), 8 waves in 1x8; wave computes 16x32
// (A-frag reused 2x). BK = 32. A rows staged via async DMA (gathered).
// GEMM1: gathered h rows x W1[e] -> relu -> act (bf16).
// GEMM2: act x W2[e] -> gate-scaled atomicAdd into residual stream x.
// ---------------------------------------------------------------------------
__global__ __launch_bounds__(256) void moe_gemm1(const bf16* __restrict__ hbf,
                                                 const int* __restrict__ rowtok,
                                                 const int* __restrict__ epad,
                                                 const int* __restrict__ ebase,
                                                 const bf16* __restrict__ W1,
                                                 const float* __restrict__ b1e,
                                                 bf16* __restrict__ act) {
  using namespace cfg;
  const int e = blockIdx.z;
  if ((int)(blockIdx.y * 16) >= epad[e]) return;
  const int row0 = ebase[e] + blockIdx.y * 16;
  const bf16* Bw = W1 + (size_t)e * Hdim * FF;
  const float* bias = b1e + e * FF;
  const int bn = blockIdx.x * 256;
  __shared__ bf16 As[16][32];
  __shared__ bf16 Bs[256][40];
  __shared__ int toks[16];
  const int tid = threadIdx.x, lane = tid & 31, wid = tid >> 5;
  const int hf = lane >> 4, nl = lane & 15;
  if (tid < 16) toks[tid] = rowtok[row0 + tid];
  v8f acc0 = {0.f, 0.f, 0.f, 0.f, 0.f, 0.f, 0.f, 0.f};
  v8f acc1 = {0.f, 0.f, 0.f, 0.f, 0.f, 0.f, 0.f, 0.f};

  for (int k0 = 0; k0 < Hdim; k0 += 32) {
    __syncthreads();
    if (tid < 64) {  // gather-stage A 16x32 via async DMA (16B per lane)
      const int flat = tid * 8, r = flat >> 5, c = flat & 31;
      async_copy_b128(&As[r][c], hbf + (size_t)toks[r] * Hdim + k0 + c);
    }
    {  // stage B 32x256 transposed
      const int idx = tid * 32, kk = idx >> 8, n0 = idx & 255;
      const bf16* s2 = Bw + (size_t)(k0 + kk) * FF + bn + n0;
#pragma unroll
      for (int i = 0; i < 32; ++i) Bs[n0 + i][kk] = s2[i];
    }
    if (k0 + 32 < Hdim) __builtin_prefetch(Bw + (size_t)(k0 + 32) * FF + bn, 0, 1);
    wait_async0();
    __syncthreads();
    const v16bf a = load_afrag(&As[nl][0], hf);
    acc0 = wmma_bf16(a, load_bfrag(&Bs[wid * 32 + nl][0], hf), acc0);
    acc1 = wmma_bf16(a, load_bfrag(&Bs[wid * 32 + 16 + nl][0], hf), acc1);
  }
#pragma unroll
  for (int r = 0; r < 8; ++r) {
    const int row = row0 + r + hf * 8;
    const int col0 = bn + wid * 32 + nl;
    act[(size_t)row * FF + col0] = (bf16)fmaxf(acc0[r] + bias[col0], 0.f);
    act[(size_t)row * FF + col0 + 16] = (bf16)fmaxf(acc1[r] + bias[col0 + 16], 0.f);
  }
}

__global__ __launch_bounds__(256) void moe_gemm2(const bf16* __restrict__ act,
                                                 const int* __restrict__ rowtok,
                                                 const float* __restrict__ rowgate,
                                                 const int* __restrict__ epad,
                                                 const int* __restrict__ ebase,
                                                 const bf16* __restrict__ W2,
                                                 const float* __restrict__ b2e,
                                                 float* __restrict__ xout) {
  using namespace cfg;
  const int e = blockIdx.z;
  if ((int)(blockIdx.y * 16) >= epad[e]) return;
  const int row0 = ebase[e] + blockIdx.y * 16;
  const bf16* Bw = W2 + (size_t)e * FF * Hdim;
  const float* bias = b2e + e * Hdim;
  const int bn = blockIdx.x * 256;
  __shared__ bf16 As[16][32];
  __shared__ bf16 Bs[256][40];
  const int tid = threadIdx.x, lane = tid & 31, wid = tid >> 5;
  const int hf = lane >> 4, nl = lane & 15;
  v8f acc0 = {0.f, 0.f, 0.f, 0.f, 0.f, 0.f, 0.f, 0.f};
  v8f acc1 = {0.f, 0.f, 0.f, 0.f, 0.f, 0.f, 0.f, 0.f};

  for (int k0 = 0; k0 < FF; k0 += 32) {
    __syncthreads();
    if (tid < 64) {  // stage A 16x32 via async DMA
      const int flat = tid * 8, r = flat >> 5, c = flat & 31;
      async_copy_b128(&As[r][c], act + (size_t)(row0 + r) * FF + k0 + c);
    }
    {  // stage B 32x256 transposed
      const int idx = tid * 32, kk = idx >> 8, n0 = idx & 255;
      const bf16* s2 = Bw + (size_t)(k0 + kk) * Hdim + bn + n0;
#pragma unroll
      for (int i = 0; i < 32; ++i) Bs[n0 + i][kk] = s2[i];
    }
    if (k0 + 32 < FF) __builtin_prefetch(Bw + (size_t)(k0 + 32) * Hdim + bn, 0, 1);
    wait_async0();
    __syncthreads();
    const v16bf a = load_afrag(&As[nl][0], hf);
    acc0 = wmma_bf16(a, load_bfrag(&Bs[wid * 32 + nl][0], hf), acc0);
    acc1 = wmma_bf16(a, load_bfrag(&Bs[wid * 32 + 16 + nl][0], hf), acc1);
  }
#pragma unroll
  for (int r = 0; r < 8; ++r) {
    const int row = row0 + r + hf * 8;
    const int tk = rowtok[row];
    const float gv = rowgate[row];
    const int col0 = bn + wid * 32 + nl;
    atomicAdd(xout + (size_t)tk * Hdim + col0, gv * (acc0[r] + bias[col0]));
    atomicAdd(xout + (size_t)tk * Hdim + col0 + 16, gv * (acc1[r] + bias[col0 + 16]));
  }
}

// ---------------------------------------------------------------------------
// Host launcher
// ---------------------------------------------------------------------------
extern "C" void kernel_launch(void* const* d_in, const int* in_sizes, int n_in,
                              void* d_out, int out_size, void* d_ws, size_t ws_size,
                              hipStream_t stream) {
  using namespace cfg;
  (void)in_sizes; (void)n_in; (void)out_size; (void)ws_size;

  const int*   ids   = (const int*)d_in[0];
  const int*   amask = (const int*)d_in[1];
  const float* tokW  = (const float*)d_in[2];
  const float* posW  = (const float*)d_in[3];
  const float* gF    = (const float*)d_in[4];
  const float* bF    = (const float*)d_in[5];

  char* cur = (char*)d_ws;
  auto take = [&](size_t bytes) { char* p = cur; cur += (bytes + 255) & ~(size_t)255; return p; };
  float* x      = (float*)take((size_t)TOK * Hdim * 4);
  bf16*  hbf    = (bf16*)take((size_t)TOK * Hdim * 2);
  bf16*  qbf    = (bf16*)take((size_t)TOK * Hdim * 2);
  bf16*  kbf    = (bf16*)take((size_t)TOK * Hdim * 2);
  bf16*  vbf    = (bf16*)take((size_t)TOK * Hdim * 2);
  bf16*  obf    = (bf16*)take((size_t)TOK * Hdim * 2);
  bf16*  wqb    = (bf16*)take((size_t)Hdim * Hdim * 2);
  bf16*  wkb    = (bf16*)take((size_t)Hdim * Hdim * 2);
  bf16*  wvb    = (bf16*)take((size_t)Hdim * Hdim * 2);
  bf16*  wob    = (bf16*)take((size_t)Hdim * Hdim * 2);
  bf16*  w1b    = (bf16*)take((size_t)NE * Hdim * FF * 2);
  bf16*  w2b    = (bf16*)take((size_t)NE * FF * Hdim * 2);
  bf16*  act    = (bf16*)take((size_t)ROWCAP * FF * 2);
  int*   cnt    = (int*)take(NE * 4);
  int*   epad   = (int*)take(NE * 4);
  int*   ebase  = (int*)take((NE + 1) * 4);
  int*   toklist  = (int*)take((size_t)NE * TOK * 4);
  float* gatelist = (float*)take((size_t)NE * TOK * 4);
  int*   rowtok   = (int*)take((size_t)ROWCAP * 4);
  float* rowgate  = (float*)take((size_t)ROWCAP * 4);

  embed_kernel<<<TOK, 256, 0, stream>>>(ids, tokW, posW, x);

  const dim3 gemmGrid(Hdim / 256, TOK / 32);
  const dim3 attnGrid(Seq / 64, NHead, Bsz);
  const dim3 moe1Grid(FF / 256, TOK / 16, NE);
  const dim3 moe2Grid(Hdim / 256, TOK / 16, NE);

  for (int l = 0; l < NL; ++l) {
    const float* const* P = (const float* const*)(d_in + 6 + l * 18);
    const float *g1 = P[0], *b1 = P[1], *Wq = P[2], *bq = P[3], *Wk = P[4], *bk = P[5];
    const float *Wv = P[6], *bv = P[7], *Wo = P[8], *bo = P[9], *g2 = P[10], *b2 = P[11];
    const float *Wg = P[12], *bg = P[13], *W1 = P[14], *b1e = P[15], *W2 = P[16], *b2e = P[17];

    const int nHH = Hdim * Hdim;
    const int nEW = NE * Hdim * FF;
    cvt_kernel<<<nHH / 256, 256, 0, stream>>>(Wq, wqb, nHH);
    cvt_kernel<<<nHH / 256, 256, 0, stream>>>(Wk, wkb, nHH);
    cvt_kernel<<<nHH / 256, 256, 0, stream>>>(Wv, wvb, nHH);
    cvt_kernel<<<nHH / 256, 256, 0, stream>>>(Wo, wob, nHH);
    cvt_kernel<<<nEW / 256, 256, 0, stream>>>(W1, w1b, nEW);
    cvt_kernel<<<nEW / 256, 256, 0, stream>>>(W2, w2b, nEW);

    // --- attention block ---
    ln_kernel<false, true><<<TOK, 256, 0, stream>>>(x, g1, b1, nullptr, hbf);
    gemm_main<false, false><<<gemmGrid, 256, 0, stream>>>(hbf, wqb, bq, nullptr, qbf, Hdim, Hdim);
    gemm_main<false, false><<<gemmGrid, 256, 0, stream>>>(hbf, wkb, bk, nullptr, kbf, Hdim, Hdim);
    gemm_main<false, false><<<gemmGrid, 256, 0, stream>>>(hbf, wvb, bv, nullptr, vbf, Hdim, Hdim);
    attention_kernel<<<attnGrid, 128, 0, stream>>>(qbf, kbf, vbf, amask, obf);
    gemm_main<false, true><<<gemmGrid, 256, 0, stream>>>(obf, wob, bo, x, nullptr, Hdim, Hdim);

    // --- MoE block (top-2 routed, not dense) ---
    ln_kernel<false, true><<<TOK, 256, 0, stream>>>(x, g2, b2, nullptr, hbf);
    hipMemsetAsync(cnt, 0, NE * sizeof(int), stream);
    gate_route<<<TOK / 8, 256, 0, stream>>>(hbf, Wg, bg, cnt, toklist, gatelist);
    finalize_route<<<1, 32, 0, stream>>>(cnt, epad, ebase);
    scatter_route<<<dim3(TOK / 256, NE), 256, 0, stream>>>(cnt, epad, ebase, toklist,
                                                           gatelist, rowtok, rowgate);
    moe_gemm1<<<moe1Grid, 256, 0, stream>>>(hbf, rowtok, epad, ebase, w1b, b1e, act);
    moe_gemm2<<<moe2Grid, 256, 0, stream>>>(act, rowtok, rowgate, epad, ebase, w2b, b2e, x);
  }

  ln_kernel<true, false><<<TOK, 256, 0, stream>>>(x, gF, bF, (float*)d_out, nullptr);
}
